// EmformerEncoder_77326591197527
// MI455X (gfx1250) — matrix operand order repaired and checked by reading, not compiled, not run on hardware
//
#include <hip/hip_runtime.h>
#include <hip/hip_bf16.h>

typedef __bf16 bf16;
typedef __attribute__((ext_vector_type(16))) __bf16 bf16x16;
typedef __attribute__((ext_vector_type(8)))  float  f32x8;

// ---------------- model constants ----------------
constexpr int Bb = 4, Dd = 512, Hh = 8, FFd = 2048, Ll = 4, Kk = 31;
constexpr int Uu = 512, RCc = 8, CHUNKc = 32, LCc = 32, MEMc = 8;
constexpr int NCH = 16, Ss = 16, Mm = 15, Rr = 128;
constexpr int Qq  = Rr + Uu + Ss;   // 656
constexpr int KVv = Mm + Rr + Uu;   // 655
constexpr int HDd = Dd / Hh;        // 64
constexpr int PEe = 2 * Uu - 1;     // 1023
constexpr int KVP = 672;            // KV padded to mult of 32
constexpr int PEP = 1024;           // PE padded
constexpr int TPAD = 544;           // 30 (left pad) + 512, padded to 16
constexpr int TOKS = Rr + Uu;       // 640
constexpr float SCALEf = 0.125f;    // HD^-0.5
constexpr float NEGINF = -1.0e8f;

// ---------------- helpers ----------------
__device__ __forceinline__ bf16 f2b(float x) {
  unsigned u = __builtin_bit_cast(unsigned, x);
  unsigned r = (u + 0x7FFFu + ((u >> 16) & 1u)) >> 16;
  unsigned short s = (unsigned short)r;
  return __builtin_bit_cast(bf16, s);
}
__device__ __forceinline__ int imax(int a, int b) { return a > b ? a : b; }
__device__ __forceinline__ int imin(int a, int b) { return a < b ? a : b; }
__device__ __forceinline__ float sigm(float x) { return 1.f / (1.f + __expf(-x)); }

// ---------------- WMMA GEMM: C[m,n] = sum_k A[m,k]*Bw[n,k] (+bias[n]) ----------------
// A fragment layout (16-bit A 16x32, ISA 7.12.2): lane = row(0..15) | +16 -> k-half,
// lane<16 holds k={0..7,16..23}, lane>=16 holds k={8..15,24..31}.
__device__ __forceinline__ bf16x16 load_a_frag(const bf16* tile, int ld) {
  int lane = threadIdx.x & 31;
  int r = lane & 15;
  int g = (lane >> 4) << 3;              // 0 or 8
  const bf16* p = tile + (size_t)r * ld + g;
  union { bf16x16 v; uint4 q[2]; } u;
  u.q[0] = *(const uint4*)(p);
  u.q[1] = *(const uint4*)(p + 16);
  return u.v;
}

constexpr int GM = 4;    // 16-row subtiles per wave (64 rows)
constexpr int GN = 2;    // 16-col subtiles per wave (32 cols)
constexpr int KCH = 512; // K-chunk staged in LDS (32 rows x 512 x bf16 = 32KB)

// B fragment (32x16 K-major) from LDS slab with row stride KCH:
// lane = col(0..15); lanes 0-15 hold K={0..7,16..23}, lanes 16-31 hold K={8..15,24..31}.
__device__ __forceinline__ bf16x16 load_b_frag_lds(const bf16* tile) {
  int lane = threadIdx.x & 31;
  int n = lane & 15;
  int g = (lane >> 4) << 3;              // 0 or 8
  const bf16* p = tile + (size_t)n * KCH + g;
  union { bf16x16 v; uint4 q[2]; } u;
  u.q[0] = *(const uint4*)(p);
  u.q[1] = *(const uint4*)(p + 16);
  return u.v;
}

__global__ __launch_bounds__(128)
void k_gemm(const bf16* __restrict__ A, const bf16* __restrict__ Bw,
            const float* __restrict__ bias, float* __restrict__ C,
            int M, int N, int Kd, int lda, int ldb, int ldc,
            long long sA, long long sB, long long sC, int sBmod, int act) {
  __shared__ alignas(16) bf16 lb[32 * KCH];

  int Mt = M >> 4;                             // total 16-row subtiles (M >= 64 always)
  int wt = blockIdx.y * 4 + threadIdx.y;       // wave macro-row (wave-uniform)
  int mt0 = wt * GM;
  if (mt0 > Mt - GM) mt0 = Mt - GM;            // clamp: overlap instead of branch
  int n0 = blockIdx.x * (GN * 16);
  int z = blockIdx.z;
  long long zb = sBmod ? (long long)(z % sBmod) : (long long)z;
  const bf16* Az = A + (long long)z * sA + (size_t)mt0 * 16 * lda;
  const bf16* Bz = Bw + zb * sB + (size_t)n0 * ldb;
  float* Cz = C + (long long)z * sC + (size_t)mt0 * 16 * ldc + n0;

  int tid = threadIdx.y * 32 + (threadIdx.x & 31);
  unsigned lds_base = (unsigned)(size_t)(void*)lb;  // low 32 bits = LDS offset

  f32x8 acc[GM][GN];
#pragma unroll
  for (int i = 0; i < GM; ++i)
#pragma unroll
    for (int j = 0; j < GN; ++j) acc[i][j] = {};

  for (int kc = 0; kc < Kd; kc += KCH) {
    int csz = Kd - kc;
    if (csz > KCH) csz = KCH;
    if (kc) __syncthreads();                   // previous chunk fully consumed

    // ---- async-stage B slab: rows n0..n0+31, cols kc..kc+csz (csz mult of 32) ----
    int cg = csz >> 3;                         // 8-elem (16B) groups per row
    int ntr = csz >> 5;                        // transfers per thread (32*cg/128)
    for (int it = 0; it < ntr; ++it) {
      int tr = tid + (it << 7);
      int row = tr / cg;
      int c8 = (tr - row * cg) << 3;
      const bf16* gp = Bz + (size_t)row * ldb + kc + c8;
      unsigned lo = lds_base + (unsigned)((row * KCH + c8) * 2);
      asm volatile("global_load_async_to_lds_b128 %0, %1, off"
                   :
                   : "v"(lo), "v"(gp)
                   : "memory");
    }
    asm volatile("s_wait_asynccnt 0" ::: "memory");
    __syncthreads();

    // ---- MAC over the staged chunk ----
    for (int k0 = 0; k0 < csz; k0 += 32) {
      bf16x16 bfr[GN];
#pragma unroll
      for (int j = 0; j < GN; ++j)
        bfr[j] = load_b_frag_lds(lb + (size_t)(j * 16) * KCH + k0);
#pragma unroll
      for (int i = 0; i < GM; ++i) {
        bf16x16 afr = load_a_frag(Az + (size_t)i * 16 * lda + kc + k0, lda);
#pragma unroll
        for (int j = 0; j < GN; ++j)
          acc[i][j] = __builtin_amdgcn_wmma_f32_16x16x32_bf16(
              false, afr, false, bfr[j], (short)0, acc[i][j], false, false);
      }
    }
  }

  int lane = threadIdx.x & 31;
  int nl = lane & 15;
  int mh = (lane >> 4) << 3;                   // C layout: vgpr r -> M=r or r+8
#pragma unroll
  for (int i = 0; i < GM; ++i) {
#pragma unroll
    for (int j = 0; j < GN; ++j) {
      float bv = bias ? bias[n0 + j * 16 + nl] : 0.f;
      float* Cs = Cz + (size_t)i * 16 * ldc + j * 16;
#pragma unroll
      for (int r = 0; r < 8; ++r) {
        float v = acc[i][j][r] + bv;
        if (act) v = v * sigm(v);              // fused SiLU
        Cs[(size_t)(r + mh) * ldc + nl] = v;
      }
    }
  }
}

// ---------------- elementwise / reduction kernels ----------------
__global__ void k_f2b(const float* __restrict__ in, bf16* __restrict__ o, long long n) {
  long long i = blockIdx.x * 256LL + threadIdx.x;
  if (i < n) o[i] = f2b(in[i]);
}
__global__ void k_zerob(bf16* __restrict__ p, long long n) {
  long long i = blockIdx.x * 256LL + threadIdx.x;
  if (i < n) p[i] = f2b(0.f);
}
__global__ void k_addscale(float* __restrict__ d, const float* __restrict__ s,
                           float a, long long n) {
  long long i = blockIdx.x * 256LL + threadIdx.x;
  if (i < n) d[i] += a * s[i];
}

__global__ __launch_bounds__(256)
void k_layernorm(const float* __restrict__ x, const float* __restrict__ g,
                 const float* __restrict__ b, float* __restrict__ y, int act) {
  int row = blockIdx.x;
  const float* xr = x + (size_t)row * Dd;
  float* yr = y + (size_t)row * Dd;
  int t = threadIdx.x;
  float v0 = xr[t], v1 = xr[t + 256];
  __shared__ float red[256];
  red[t] = v0 + v1;
  __syncthreads();
#pragma unroll
  for (int o = 128; o > 0; o >>= 1) { if (t < o) red[t] += red[t + o]; __syncthreads(); }
  float mu = red[0] * (1.f / Dd);
  __syncthreads();
  float d0 = v0 - mu, d1 = v1 - mu;
  red[t] = d0 * d0 + d1 * d1;
  __syncthreads();
#pragma unroll
  for (int o = 128; o > 0; o >>= 1) { if (t < o) red[t] += red[t + o]; __syncthreads(); }
  float inv = rsqrtf(red[0] * (1.f / Dd) + 1e-5f);
  float r0 = d0 * inv * g[t] + b[t];
  float r1 = d1 * inv * g[t + 256] + b[t + 256];
  if (act) { r0 *= sigm(r0); r1 *= sigm(r1); }
  yr[t] = r0;
  yr[t + 256] = r1;
}

// rows are (token*B + b); average 32 tokens per chunk
__global__ __launch_bounds__(256)
void k_chunkavg(const float* __restrict__ a, float* __restrict__ s) {
  int cb = blockIdx.x;
  int c = cb / Bb, b = cb % Bb;
  for (int d = threadIdx.x; d < Dd; d += 256) {
    float acc = 0.f;
    for (int tt = 0; tt < CHUNKc; ++tt)
      acc += a[((size_t)((c * CHUNKc + tt) * Bb + b)) * Dd + d];
    s[((size_t)(c * Bb + b)) * Dd + d] = acc * (1.f / CHUNKc);
  }
}

// src = [rc(128); utterance(512)] gathered from x (520,B,D)
__global__ void k_init(const float* __restrict__ x, float* __restrict__ src) {
  long long i = blockIdx.x * 256LL + threadIdx.x;
  long long tot = (long long)TOKS * Bb * Dd;
  if (i >= tot) return;
  int d = (int)(i % Dd);
  long long rb = i / Dd;
  int b = (int)(rb % Bb);
  int tok = (int)(rb / Bb);
  int xt;
  if (tok < Rr) {
    int sgm = tok >> 3, r = tok & 7;
    xt = (sgm < NCH - 1) ? (sgm + 1) * CHUNKc + r : Uu + r;  // RC_IDX
  } else xt = tok - Rr;
  src[i] = x[((size_t)xt * Bb + b) * Dd + d];
}

// sinusoidal positional embedding, bf16, rows padded to PEP
__global__ void k_posemb(bf16* __restrict__ pe) {
  long long i = blockIdx.x * 256LL + threadIdx.x;
  long long tot = (long long)PEP * Dd;
  if (i >= tot) return;
  int d = (int)(i % Dd);
  int p = (int)(i / Dd);
  float v = 0.f;
  if (p < PEe) {
    float posv = (float)(Uu - 1 - p);
    float div = __expf(-(float)(d & ~1) * (9.210340371976184f / (float)Dd)); // ln(1e4)
    float a = posv * div;
    v = (d & 1) ? cosf(a) : sinf(a);
  }
  pe[i] = f2b(v);
}

// pos fp32 [PEP,D] -> posh bf16 [H,PEP,HD]
__global__ void k_posprep(const float* __restrict__ pos, bf16* __restrict__ posh) {
  long long i = blockIdx.x * 256LL + threadIdx.x;
  long long tot = (long long)Hh * PEP * HDd;
  if (i >= tot) return;
  int dd = (int)(i % HDd);
  long long r = i / HDd;
  int p = (int)(r % PEP);
  int h = (int)(r / PEP);
  float v = (p < PEe) ? pos[(size_t)p * Dd + h * HDd + dd] : 0.f;
  posh[i] = f2b(v);
}

// query fp32 [Q*B, D] -> qh = q*SCALE+pbu as [B,H,Q,HD]; qpos = q*SCALE+pbv as [B,H,U,HD]
__global__ void k_qprep(const float* __restrict__ query, const float* __restrict__ pbu,
                        const float* __restrict__ pbv, bf16* __restrict__ qh,
                        bf16* __restrict__ qpos) {
  long long i = blockIdx.x * 256LL + threadIdx.x;
  long long tot = (long long)Qq * Bb * Dd;
  if (i >= tot) return;
  int d = (int)(i % Dd);
  long long rb = i / Dd;
  int b = (int)(rb % Bb);
  int q = (int)(rb / Bb);
  int h = d >> 6, dd = d & 63;
  float qv = query[i] * SCALEf;
  qh[(((size_t)(b * Hh + h)) * Qq + q) * HDd + dd] = f2b(qv + pbu[d]);
  int u = q - Rr;
  if (u >= 0 && u < Uu)
    qpos[(((size_t)(b * Hh + h)) * Uu + u) * HDd + dd] = f2b(qv + pbv[d]);
}

// kv fp32 [KV*B rows (padded), 2D] -> kh [B,H,KVP,HD], vh [B,H,HD,KVP] (zero padded)
__global__ void k_kvprep(const float* __restrict__ kv, bf16* __restrict__ kh,
                         bf16* __restrict__ vh) {
  long long i = blockIdx.x * 256LL + threadIdx.x;
  long long tot = (long long)Bb * Hh * KVP * HDd;
  if (i >= tot) return;
  int dd = (int)(i % HDd);
  long long r = i / HDd;
  int kk = (int)(r % KVP);
  int bh = (int)(r / KVP);
  int h = bh % Hh, b = bh / Hh;
  float kval = 0.f, vval = 0.f;
  if (kk < KVv) {
    const float* row = kv + ((size_t)kk * Bb + b) * (2 * Dd);
    kval = row[h * HDd + dd];
    vval = row[Dd + h * HDd + dd];
  }
  kh[((size_t)bh * KVP + kk) * HDd + dd] = f2b(kval);
  vh[((size_t)bh * HDd + dd) * KVP + kk] = f2b(vval);
}

// chunked attention mask (allowed = !ATTN_MASK)
__device__ __forceinline__ bool attn_allowed(int q, int k) {
  int ci; bool has_rc;
  if (q < Rr)           { ci = q >> 3;        has_rc = true;  }
  else if (q < Rr + Uu) { ci = (q - Rr) >> 5; has_rc = true;  }
  else                  { ci = q - (Rr + Uu); has_rc = false; }
  if (k < Mm) { int ms = imax(ci - MEMc, 0); return (k >= ms) && (k < ci); }
  if (k < Mm + Rr) {
    if (!has_rc) return false;
    int kr = k - Mm;
    return (kr >= ci * RCc) && (kr < ci * RCc + RCc);
  }
  int ku = k - (Mm + Rr);
  int cs = imax(ci * CHUNKc - LCc, 0);
  int ce = imin((ci + 1) * CHUNKc, Uu);
  return (ku >= cs) && (ku < ce);
}

// scores = ac + rel_shift(bd) with masks, softmax over KV, write bf16 probs [B,H,Q,KVP]
__global__ __launch_bounds__(128)
void k_scores(const float* __restrict__ ac, const float* __restrict__ bd,
              const int* __restrict__ lengths, bf16* __restrict__ probs) {
  int q = blockIdx.x, h = blockIdx.y, b = blockIdx.z;
  size_t bh = (size_t)(b * Hh + h);
  const float* acr = ac + (bh * Qq + q) * KVP;
  bf16* pr = probs + (bh * Qq + q) * KVP;
  int ol = imin(imax(lengths[b] - RCc, 0), Uu);
  int kpad0 = KVv - Uu + ol;
  int u = q - Rr;
  bool hasbd = (u >= 0 && u < Uu);
  const float* bdr = bd + (bh * Uu + (hasbd ? u : 0)) * PEP;
  __shared__ float red[128];
  int t = threadIdx.x;
  float w[6];
  float mx = -3.4e38f;
#pragma unroll
  for (int i = 0; i < 6; ++i) {
    int k = t + i * 128;
    float v = -3.4e38f;
    if (k < KVv) {
      v = acr[k];
      if (hasbd && k >= Mm + Rr) v += bdr[(Uu - 1 - u) + (k - (Mm + Rr))]; // rel_shift
      if (!attn_allowed(q, k) || k >= kpad0) v = NEGINF;
    }
    w[i] = v;
    mx = fmaxf(mx, v);
  }
  red[t] = mx; __syncthreads();
  for (int o = 64; o > 0; o >>= 1) { if (t < o) red[t] = fmaxf(red[t], red[t + o]); __syncthreads(); }
  mx = red[0]; __syncthreads();
  float sum = 0.f;
#pragma unroll
  for (int i = 0; i < 6; ++i) {
    int k = t + i * 128;
    float e = (k < KVv) ? __expf(w[i] - mx) : 0.f;
    w[i] = e; sum += e;
  }
  red[t] = sum; __syncthreads();
  for (int o = 64; o > 0; o >>= 1) { if (t < o) red[t] += red[t + o]; __syncthreads(); }
  float inv = 1.f / red[0];
#pragma unroll
  for (int i = 0; i < 6; ++i) {
    int k = t + i * 128;
    if (k < KVP) pr[k] = f2b((k < KVv) ? w[i] * inv : 0.f);
  }
}

// attn [B,H,Q,HD] fp32 -> bf16 rows [(q*B+b), D]
__global__ void k_attnrows(const float* __restrict__ attn, bf16* __restrict__ arows) {
  long long i = blockIdx.x * 256LL + threadIdx.x;
  long long tot = (long long)Qq * Bb * Dd;
  if (i >= tot) return;
  int d = (int)(i % Dd);
  long long rb = i / Dd;
  int b = (int)(rb % Bb);
  int q = (int)(rb / Bb);
  int h = d >> 6, dd = d & 63;
  arows[i] = f2b(attn[(((size_t)(b * Hh + h)) * Qq + q) * HDd + dd]);
}

// src[:640B rows] += attn_out; mem_next = tanh(summary rows)[:15]
__global__ void k_attnapply(const float* __restrict__ ao, float* __restrict__ src,
                            float* __restrict__ memn) {
  long long i = blockIdx.x * 256LL + threadIdx.x;
  long long tot = (long long)Qq * Bb * Dd;
  if (i >= tot) return;
  long long lim = (long long)TOKS * Bb * Dd;
  if (i < lim) src[i] += ao[i];
  else {
    long long j = i - lim;
    if (j < (long long)Mm * Bb * Dd) memn[j] = tanhf(ao[i]);
  }
}

// GLU from pw1 output h1 [(t*B+b), 1024] into glu [B,D,TPAD], left pad uses bias
__global__ void k_glu(const float* __restrict__ h1, const float* __restrict__ b1,
                      float* __restrict__ glu) {
  long long i = blockIdx.x * 256LL + threadIdx.x;
  long long tot = (long long)Bb * Dd * TPAD;
  if (i >= tot) return;
  int tt = (int)(i % TPAD);
  long long r = i / TPAD;
  int d = (int)(r % Dd);
  int b = (int)(r / Dd);
  float v = 0.f;
  if (tt < Kk - 1) {
    v = b1[d] * sigm(b1[Dd + d]);
  } else if (tt < Kk - 1 + Uu) {
    int t = tt - (Kk - 1);
    const float* row = h1 + ((size_t)t * Bb + b) * (2 * Dd);
    v = row[d] * sigm(row[Dd + d]);
  }
  glu[i] = v;
}

// depthwise conv (cross-correlation, VALID over TPAD) -> rows [(t*B+b), D]
__global__ void k_dwconv(const float* __restrict__ glu, const float* __restrict__ w,
                         const float* __restrict__ bias, float* __restrict__ out) {
  long long i = blockIdx.x * 256LL + threadIdx.x;
  long long tot = (long long)Uu * Bb * Dd;
  if (i >= tot) return;
  int d = (int)(i % Dd);
  long long r = i / Dd;
  int b = (int)(r % Bb);
  int t = (int)(r / Bb);
  const float* g = glu + ((size_t)b * Dd + d) * TPAD + t;
  float acc = bias[d];
#pragma unroll
  for (int j = 0; j < Kk; ++j) acc += w[d * Kk + j] * g[j];
  out[i] = acc;
}

// ---------------- host orchestration ----------------
extern "C" void kernel_launch(void* const* d_in, const int* in_sizes, int n_in,
                              void* d_out, int out_size, void* d_ws, size_t ws_size,
                              hipStream_t stream) {
  (void)in_sizes; (void)n_in; (void)out_size; (void)ws_size;
  const float* x = (const float*)d_in[0];
  const int* lengths = (const int*)d_in[1];

  char* base = (char*)d_ws;
  size_t off = 0;
  auto alloc = [&](size_t bytes) -> char* {
    char* p = base + off;
    off += (bytes + 255) & ~(size_t)255;
    return p;
  };

  float* src   = (float*)alloc((size_t)TOKS * Bb * Dd * 4);
  float* lnbuf = (float*)alloc((size_t)TOKS * Bb * Dd * 4);
  float* summ  = (float*)alloc((size_t)NCH * Bb * Dd * 4);
  float* memA  = (float*)alloc((size_t)Mm * Bb * Dd * 4);
  float* memB  = (float*)alloc((size_t)Mm * Bb * Dd * 4);
  float* hbig  = (float*)alloc((size_t)TOKS * Bb * FFd * 4);
  bf16*  abf   = (bf16*)alloc((size_t)2624 * FFd * 2);
  bf16*  wbf   = (bf16*)alloc((size_t)FFd * Dd * 2);
  float* qbuf  = (float*)alloc((size_t)Qq * Bb * Dd * 4);
  float* kvbuf = (float*)alloc((size_t)2624 * 2 * Dd * 4);
  bf16*  qh    = (bf16*)alloc((size_t)Bb * Hh * Qq * HDd * 2);
  bf16*  qpos  = (bf16*)alloc((size_t)Bb * Hh * Uu * HDd * 2);
  bf16*  kh    = (bf16*)alloc((size_t)Bb * Hh * KVP * HDd * 2);
  bf16*  vh    = (bf16*)alloc((size_t)Bb * Hh * HDd * KVP * 2);
  bf16*  pebf  = (bf16*)alloc((size_t)PEP * Dd * 2);
  float* posf  = (float*)alloc((size_t)PEP * Dd * 4);
  bf16*  posh  = (bf16*)alloc((size_t)Hh * PEP * HDd * 2);
  float* ac    = (float*)alloc((size_t)Bb * Hh * Qq * KVP * 4);
  float* bdb   = (float*)alloc((size_t)Bb * Hh * Uu * PEP * 4);
  bf16*  probs = (bf16*)alloc((size_t)Bb * Hh * Qq * KVP * 2);
  float* attn  = (float*)alloc((size_t)Bb * Hh * Qq * HDd * 4);
  bf16*  arows = (bf16*)alloc((size_t)Qq * Bb * Dd * 2);
  float* aout  = (float*)alloc((size_t)Qq * Bb * Dd * 4);
  float* glu   = (float*)alloc((size_t)Bb * Dd * TPAD * 4);
  float* convt = (float*)alloc((size_t)Uu * Bb * Dd * 4);

  auto EW = [&](long long n) { return dim3((unsigned)((n + 255) / 256)); };
  auto cvt = [&](const float* in, bf16* o, long long n) {
    k_f2b<<<EW(n), 256, 0, stream>>>(in, o, n);
  };
  auto gemm = [&](const bf16* A, const bf16* Bw, const float* bias, float* C,
                  int M, int N, int Kd, int lda, int ldb, int ldc,
                  long long sA, long long sB, long long sC, int sBmod, int batch, int act) {
    int Mt = M >> 4;                          // 16-row subtiles
    int waveRows = (Mt + GM - 1) / GM;        // 64-row wave macro-rows
    dim3 g(N / (GN * 16), (waveRows + 3) / 4, batch);
    dim3 blk(32, 4, 1);
    k_gemm<<<g, blk, 0, stream>>>(A, Bw, bias, C, M, N, Kd, lda, ldb, ldc,
                                  sA, sB, sC, sBmod, act);
  };

  // ---- initial state ----
  k_init<<<EW((long long)TOKS * Bb * Dd), 256, 0, stream>>>(x, src);
  k_chunkavg<<<NCH * Bb, 256, 0, stream>>>(src + (size_t)Rr * Bb * Dd, summ);
  hipMemcpyAsync(memA, summ, (size_t)Mm * Bb * Dd * 4, hipMemcpyDeviceToDevice, stream);
  k_posemb<<<EW((long long)PEP * Dd), 256, 0, stream>>>(pebf);

  float* mem = memA;
  float* memn = memB;

  for (int l = 0; l < Ll; ++l) {
    auto PL = [&](int idx, size_t st) -> const float* {
      return (const float*)d_in[idx] + (size_t)l * st;
    };
    const float* ffmac_w1 = PL(2, (size_t)FFd * Dd);
    const float* ffmac_b1 = PL(3, FFd);
    const float* ffmac_w2 = PL(4, (size_t)Dd * FFd);
    const float* ffmac_b2 = PL(5, Dd);
    const float* ff_w1 = PL(6, (size_t)FFd * Dd);
    const float* ff_b1 = PL(7, FFd);
    const float* ff_w2 = PL(8, (size_t)Dd * FFd);
    const float* ff_b2 = PL(9, Dd);
    const float* q_w = PL(10, (size_t)Dd * Dd);
    const float* q_b = PL(11, Dd);
    const float* kv_w = PL(12, (size_t)2 * Dd * Dd);
    const float* kv_b = PL(13, 2 * Dd);
    const float* out_w = PL(14, (size_t)Dd * Dd);
    const float* out_b = PL(15, Dd);
    const float* pos_w = PL(16, (size_t)Dd * Dd);
    const float* pbu = PL(17, Dd);
    const float* pbv = PL(18, Dd);
    const float* pw1_w = PL(19, (size_t)2 * Dd * Dd);
    const float* pw1_b = PL(20, 2 * Dd);
    const float* dw_w = PL(21, (size_t)Dd * Kk);
    const float* dw_b = PL(22, Dd);
    const float* pw2_w = PL(23, (size_t)Dd * Dd);
    const float* pw2_b = PL(24, Dd);
    const float* cn_g = PL(25, Dd);
    const float* cn_b = PL(26, Dd);
    const float* ln0_g = PL(27, Dd); const float* ln0_b = PL(28, Dd);
    const float* ln1_g = PL(29, Dd); const float* ln1_b = PL(30, Dd);
    const float* ln2_g = PL(31, Dd); const float* ln2_b = PL(32, Dd);
    const float* ln3_g = PL(33, Dd); const float* ln3_b = PL(34, Dd);
    const float* ln4_g = PL(35, Dd); const float* ln4_b = PL(36, Dd);

    const int rows = TOKS * Bb;  // 2560

    auto ffn = [&](const float* w1, const float* b1, const float* w2, const float* b2,
                   const float* lg, const float* lb2) {
      k_layernorm<<<rows, 256, 0, stream>>>(src, lg, lb2, lnbuf, 0);
      cvt(lnbuf, abf, (long long)rows * Dd);
      cvt(w1, wbf, (long long)FFd * Dd);
      gemm(abf, wbf, b1, hbig, rows, FFd, Dd, Dd, Dd, FFd, 0, 0, 0, 0, 1, 1); // SiLU fused
      cvt(hbig, abf, (long long)rows * FFd);
      cvt(w2, wbf, (long long)Dd * FFd);
      gemm(abf, wbf, b2, lnbuf, rows, Dd, FFd, FFd, FFd, Dd, 0, 0, 0, 0, 1, 0);
      k_addscale<<<EW((long long)rows * Dd), 256, 0, stream>>>(src, lnbuf, 0.5f,
                                                               (long long)rows * Dd);
    };

    // ---- macaron FFN ----
    ffn(ffmac_w1, ffmac_b1, ffmac_w2, ffmac_b2, ln0_g, ln0_b);

    // ---- attention ----
    k_layernorm<<<rows, 256, 0, stream>>>(src, ln1_g, ln1_b, lnbuf, 0);
    k_chunkavg<<<NCH * Bb, 256, 0, stream>>>(lnbuf + (size_t)Rr * Bb * Dd, summ);

    // q_in = [rc; utt; summary]  (exactly Q*B = 2624 rows)
    cvt(lnbuf, abf, (long long)rows * Dd);
    cvt(summ, abf + (size_t)rows * Dd, (long long)NCH * Bb * Dd);
    cvt(q_w, wbf, (long long)Dd * Dd);
    gemm(abf, wbf, q_b, qbuf, Qq * Bb, Dd, Dd, Dd, Dd, Dd, 0, 0, 0, 0, 1, 0);

    // kv_in = [memory; rc; utt] (2620 rows, zero-padded to 2624)
    cvt(mem, abf, (long long)Mm * Bb * Dd);
    cvt(lnbuf, abf + (size_t)Mm * Bb * Dd, (long long)rows * Dd);
    k_zerob<<<EW((long long)4 * Dd), 256, 0, stream>>>(abf + (size_t)KVv * Bb * Dd,
                                                       (long long)4 * Dd);
    cvt(kv_w, wbf, (long long)2 * Dd * Dd);
    gemm(abf, wbf, kv_b, kvbuf, 2624, 2 * Dd, Dd, Dd, Dd, 2 * Dd, 0, 0, 0, 0, 1, 0);

    k_qprep<<<EW((long long)Qq * Bb * Dd), 256, 0, stream>>>(qbuf, pbu, pbv, qh, qpos);
    k_kvprep<<<EW((long long)Bb * Hh * KVP * HDd), 256, 0, stream>>>(kvbuf, kh, vh);

    // pos = POS_EMB @ pos_w.T, repacked per-head
    cvt(pos_w, wbf, (long long)Dd * Dd);
    gemm(pebf, wbf, nullptr, posf, PEP, Dd, Dd, Dd, Dd, Dd, 0, 0, 0, 0, 1, 0);
    k_posprep<<<EW((long long)Hh * PEP * HDd), 256, 0, stream>>>(posf, posh);

    // ac[b,h,q,k] = (q*SCALE+pbu) . k   (batched over B*H)
    gemm(qh, kh, nullptr, ac, Qq, KVP, HDd, HDd, HDd, KVP,
         (long long)Qq * HDd, (long long)KVP * HDd, (long long)Qq * KVP, 0, Bb * Hh, 0);
    // bd[b,h,u,p] = (q*SCALE+pbv) . pos  (B shared across b via sBmod=H)
    gemm(qpos, posh, nullptr, bdb, Uu, PEP, HDd, HDd, HDd, PEP,
         (long long)Uu * HDd, (long long)PEP * HDd, (long long)Uu * PEP, Hh, Bb * Hh, 0);

    {
      dim3 sg(Qq, Hh, Bb);
      k_scores<<<sg, 128, 0, stream>>>(ac, bdb, lengths, probs);
    }

    // attn = probs @ v
    gemm(probs, vh, nullptr, attn, Qq, HDd, KVP, KVP, KVP, HDd,
         (long long)Qq * KVP, (long long)HDd * KVP, (long long)Qq * HDd, 0, Bb * Hh, 0);

    k_attnrows<<<EW((long long)Qq * Bb * Dd), 256, 0, stream>>>(attn, arows);
    cvt(out_w, wbf, (long long)Dd * Dd);
    gemm(arows, wbf, out_b, aout, Qq * Bb, Dd, Dd, Dd, Dd, Dd, 0, 0, 0, 0, 1, 0);
    k_attnapply<<<EW((long long)Qq * Bb * Dd), 256, 0, stream>>>(aout, src, memn);

    // ---- conv module ----
    k_layernorm<<<rows, 256, 0, stream>>>(src, ln2_g, ln2_b, lnbuf, 0);
    cvt(lnbuf + (size_t)Rr * Bb * Dd, abf, (long long)Uu * Bb * Dd);
    cvt(pw1_w, wbf, (long long)2 * Dd * Dd);
    gemm(abf, wbf, pw1_b, hbig, Uu * Bb, 2 * Dd, Dd, Dd, Dd, 2 * Dd, 0, 0, 0, 0, 1, 0);
    k_glu<<<EW((long long)Bb * Dd * TPAD), 256, 0, stream>>>(hbig, pw1_b, glu);
    k_dwconv<<<EW((long long)Uu * Bb * Dd), 256, 0, stream>>>(glu, dw_w, dw_b, convt);
    k_layernorm<<<Uu * Bb, 256, 0, stream>>>(convt, cn_g, cn_b, lnbuf, 1); // LN + SiLU
    cvt(lnbuf, abf, (long long)Uu * Bb * Dd);
    cvt(pw2_w, wbf, (long long)Dd * Dd);
    gemm(abf, wbf, pw2_b, convt, Uu * Bb, Dd, Dd, Dd, Dd, Dd, 0, 0, 0, 0, 1, 0);
    k_addscale<<<EW((long long)Uu * Bb * Dd), 256, 0, stream>>>(
        src + (size_t)Rr * Bb * Dd, convt, 1.0f, (long long)Uu * Bb * Dd);

    // ---- final FFN + LN ----
    ffn(ff_w1, ff_b1, ff_w2, ff_b2, ln3_g, ln3_b);
    k_layernorm<<<rows, 256, 0, stream>>>(src, ln4_g, ln4_b, src, 0);

    float* t = mem; mem = memn; memn = t;
  }

  hipMemcpyAsync(d_out, src + (size_t)Rr * Bb * Dd, (size_t)Uu * Bb * Dd * sizeof(float),
                 hipMemcpyDeviceToDevice, stream);
}
